// MultiheadAttention_5377299054778
// MI455X (gfx1250) — matrix operand order repaired and checked by reading, not compiled, not run on hardware
//
#include <hip/hip_runtime.h>
#include <stdint.h>

// ---------------------------------------------------------------------------
// MI455X (gfx1250) causal multi-head attention, bf16 WMMA pipeline with
// async global->LDS staging (ASYNCcnt) for GEMM tiles and flash Q/K tiles.
//   B=4, T=2048, C=1024, H=16, D=64
// Workspace layout (bf16 shorts), ~89 MB total:
//   xb   : M*C        (x in bf16, row-major [M][K])
//   wab  : 3C*C       (W_attn^T in bf16, [N][K])
//   wpb  : C*C        (W_proj^T in bf16, [N][K])
//   qkvb : M*3C       (qkv = x@W_attn+b in bf16)
//   yb   : M*C        (attention output in bf16)
// ---------------------------------------------------------------------------

#define CDIM   1024
#define N3DIM  3072
#define NHEADS 16
#define HDIM   64
#define SEQT   2048
#define NBATCH 4

typedef __attribute__((ext_vector_type(16))) __bf16 bf16x16;
typedef __attribute__((ext_vector_type(8)))  float  f32x8;

union FragBF { uint4 q[2]; bf16x16 v; };
union AccF   { f32x8 v; float f[8]; };

__device__ __forceinline__ unsigned short f32_to_bf16(float f) {
  uint32_t u = __builtin_bit_cast(uint32_t, f);
  u += 0x7fffu + ((u >> 16) & 1u);   // round-to-nearest-even
  return (unsigned short)(u >> 16);
}

// Async DMA: global -> LDS, 128 bits per lane, tracked by ASYNCcnt.
// LDS byte address = low 32 bits of the generic pointer (LDS generic addresses
// are {SHARED_BASE aperture | lds_offset}, ISA 10.2 aperture rules).
__device__ __forceinline__ void async_copy_b128(void* lds_dst, const void* gsrc) {
  uint32_t ldsa = (uint32_t)(uintptr_t)lds_dst;
  uint64_t ga   = (uint64_t)(uintptr_t)gsrc;
  asm volatile("global_load_async_to_lds_b128 %0, %1, off"
               :: "v"(ldsa), "v"(ga) : "memory");
}
__device__ __forceinline__ void async_wait0() {
  asm volatile("s_wait_asynccnt 0" ::: "memory");
}

// A-fragment (16 rows x 32 K, 16-bit). LDS layout [row][k contiguous],
// rowStride in shorts (16B-aligned). ISA 7.12.2:
// lanes 0-15 row=lane, VGPR0-3 K=0..7, VGPR4-7 K=16..23;
// lanes 16-31 row=lane-16, K=8..15 / 24..31.
__device__ __forceinline__ bf16x16 frag_a(const unsigned short* base, int rowStride, int koff) {
  const int lane = threadIdx.x & 31;
  const int r = lane & 15, hi = lane >> 4;
  const unsigned short* p = base + r * rowStride + koff + hi * 8;
  FragBF f;
  f.q[0] = *(const uint4*)(p);        // K = koff + hi*8 .. +7
  f.q[1] = *(const uint4*)(p + 16);   // K = koff + 16 + hi*8 .. +7
  return f.v;
}

// B-fragment (32 K x 16 cols, 16-bit), stored K-major in LDS: [n][k].
// Lanes 0-15: col=lane, K=0..7 / 16..23; lanes 16-31: col=lane-16, K=8..15 / 24..31.
__device__ __forceinline__ bf16x16 frag_b(const unsigned short* base, int rowStride, int koff) {
  const int lane = threadIdx.x & 31;
  const int n = lane & 15, hi = lane >> 4;
  const unsigned short* p = base + n * rowStride + koff + hi * 8;
  FragBF f;
  f.q[0] = *(const uint4*)(p);        // K = koff + hi*8 .. +7
  f.q[1] = *(const uint4*)(p + 16);   // K = koff + 16 + hi*8 .. +7
  return f.v;
}

__device__ __forceinline__ f32x8 wmma_bf16f32(bf16x16 a, bf16x16 b, f32x8 c) {
  return __builtin_amdgcn_wmma_f32_16x16x32_bf16(false, a, false, b, (short)0, c, false, false);
}

// ---------------------------------------------------------------------------
__global__ __launch_bounds__(256)
void cvt_f32_to_bf16(const float* __restrict__ in, unsigned short* __restrict__ out, int n4) {
  int i = blockIdx.x * blockDim.x + threadIdx.x;
  if (i >= n4) return;
  float4 v = *(const float4*)(in + (size_t)i * 4);
  unsigned short* o = out + (size_t)i * 4;
  o[0] = f32_to_bf16(v.x); o[1] = f32_to_bf16(v.y);
  o[2] = f32_to_bf16(v.z); o[3] = f32_to_bf16(v.w);
}

// out_bf16t[n][k] = in_f32[k][n]  (weights transposed once so GEMM B-tiles
// stage as pure b128 async copies in fragment layout).
__global__ __launch_bounds__(256)
void transpose_cvt_f32_to_bf16(const float* __restrict__ in,
                               unsigned short* __restrict__ out,
                               int K, int N) {
  __shared__ float t[32][33];
  const int tx = threadIdx.x & 31, ty = threadIdx.x >> 5;   // 32 x 8
  const int nb = blockIdx.x * 32, kb = blockIdx.y * 32;
#pragma unroll
  for (int j = ty; j < 32; j += 8)
    t[j][tx] = in[(size_t)(kb + j) * N + nb + tx];
  __syncthreads();
#pragma unroll
  for (int j = ty; j < 32; j += 8)
    out[(size_t)(nb + j) * K + kb + tx] = f32_to_bf16(t[tx][j]);
}

// ---------------------------------------------------------------------------
// C(MxN) = A(MxK,bf16) @ Bt^T + bias(N), Bt given transposed [N][K] bf16.
// 128x128 tile, 256 thr (8 waves), each wave 32(M) x 64(N) WMMA tiles, K by 32.
// OUT_BF16 selects bf16 vs f32 result stores at compile time.
template <bool OUT_BF16>
__global__ __launch_bounds__(256)
void gemm_bias_wmma(const unsigned short* __restrict__ A,    // [M][K]
                    const unsigned short* __restrict__ Bt,   // [N][K]
                    const float* __restrict__ bias,
                    void* __restrict__ Cout,
                    int M, int N, int K) {
  __shared__ unsigned short lA[128 * 40];   // [m][k], stride 40 (80B, 16B-aligned)
  __shared__ unsigned short lB[128 * 40];   // [n][k]
  const int tid = threadIdx.x;
  const int lane = tid & 31, w = tid >> 5;
  const int wr = w >> 1, wc = w & 1;
  const int mbase = blockIdx.y * 128, nbase = blockIdx.x * 128;

  AccF acc[2][4];
#pragma unroll
  for (int i = 0; i < 2; ++i)
#pragma unroll
    for (int j = 0; j < 4; ++j)
#pragma unroll
      for (int r = 0; r < 8; ++r) acc[i][j].f[r] = 0.0f;

  const int srow = tid >> 2, sc8 = (tid & 3) * 8;   // 256 threads cover half a tile

  for (int k0 = 0; k0 < K; k0 += 32) {
    __syncthreads();   // prior iteration's fragment reads are complete
    // Stage A tile 128x32 and B tile 128x32 via async DMA (4 b128/thread).
#pragma unroll
    for (int i = 0; i < 2; ++i) {
      int row = srow + i * 64;
      async_copy_b128(&lA[row * 40 + sc8], A  + (size_t)(mbase + row) * K + k0 + sc8);
      async_copy_b128(&lB[row * 40 + sc8], Bt + (size_t)(nbase + row) * K + k0 + sc8);
    }
    async_wait0();
    __syncthreads();

    bf16x16 afr[2], bfr[4];
#pragma unroll
    for (int i = 0; i < 2; ++i) afr[i] = frag_a(&lA[(wr * 32 + i * 16) * 40], 40, 0);
#pragma unroll
    for (int j = 0; j < 4; ++j) bfr[j] = frag_b(&lB[(wc * 64 + j * 16) * 40], 40, 0);
#pragma unroll
    for (int i = 0; i < 2; ++i)
#pragma unroll
      for (int j = 0; j < 4; ++j)
        acc[i][j].v = wmma_bf16f32(afr[i], bfr[j], acc[i][j].v);
  }

  // Epilogue: C layout — VGPR r: M=r+8*hi, N=lane&15.
  // One base offset per accumulator tile, stepped by N per row (no per-element
  // 64-bit multiplies, no runtime output-type branch).
  const int cn = lane & 15, hi = lane >> 4;
#pragma unroll
  for (int i = 0; i < 2; ++i)
#pragma unroll
    for (int j = 0; j < 4; ++j) {
      const int gn = nbase + wc * 64 + j * 16 + cn;
      const int gm0 = mbase + wr * 32 + i * 16 + hi * 8;
      const float bv = bias[gn];
      size_t off = (size_t)gm0 * N + gn;
#pragma unroll
      for (int r = 0; r < 8; ++r, off += N) {
        float val = acc[i][j].f[r] + bv;
        if (OUT_BF16) ((unsigned short*)Cout)[off] = f32_to_bf16(val);
        else          ((float*)Cout)[off] = val;
      }
    }
}

// ---------------------------------------------------------------------------
// Flash attention: one WG = 64 q-rows of one (b,h); 4 waves x 16 rows.
// Streams K/V in 32-key chunks; S and PV via bf16 WMMA; softmax online in f32.
// Q/K staged with async DMA; V staged with a transpose scatter.
__global__ __launch_bounds__(128)
void flash_attn_wmma(const unsigned short* __restrict__ qkv,  // (B*T) x 3072 bf16
                     unsigned short* __restrict__ y) {        // (B*T) x 1024 bf16
  __shared__ unsigned short lQ[64 * 72];      // [qrow][d]
  __shared__ unsigned short lK[32 * 72];      // [key][d]
  __shared__ unsigned short lVt[64 * 40];     // transposed: [d][key]
  __shared__ unsigned short lP[4][16 * 40];   // per-wave P stage: [qrow][key]

  const int tid = threadIdx.x, lane = tid & 31, w = tid >> 5;
  const int qt = blockIdx.x, h = blockIdx.y, b = blockIdx.z;
  const int q0 = qt * 64;
  const size_t rs = N3DIM;
  const size_t baseQ = ((size_t)b * SEQT) * rs + h * HDIM;
  const size_t baseK = baseQ + CDIM;
  const size_t baseV = baseQ + 2 * CDIM;

  // Async-load Q tile 64x64 (512 b128 / 128 threads).
#pragma unroll
  for (int i = 0; i < 4; ++i) {
    int idx = tid + i * 128;
    int row = idx >> 3, c8 = (idx & 7) * 8;
    async_copy_b128(&lQ[row * 72 + c8],
                    qkv + baseQ + (size_t)(q0 + row) * rs + c8);
  }

  const int cn = lane & 15, hi = lane >> 4;
  float row_max[8], row_sum[8];
  AccF o[4];
#pragma unroll
  for (int r = 0; r < 8; ++r) { row_max[r] = -3.0e38f; row_sum[r] = 0.0f; }
#pragma unroll
  for (int d = 0; d < 4; ++d)
#pragma unroll
    for (int r = 0; r < 8; ++r) o[d].f[r] = 0.0f;

  const float scale = 0.125f;  // 1/sqrt(64)
  const int nchunks = (q0 + 64) / 32;

  for (int kt = 0; kt < nchunks; ++kt) {
    __syncthreads();
    // Async-stage K chunk 32x64 (2 b128 / thread).
#pragma unroll
    for (int i = 0; i < 2; ++i) {
      int idx = tid + i * 128;
      int row = idx >> 3, c8 = (idx & 7) * 8;
      async_copy_b128(&lK[row * 72 + c8],
                      qkv + baseK + (size_t)(kt * 32 + row) * rs + c8);
    }
    // Stage V chunk transposed: [d][key] (needs element scatter).
#pragma unroll
    for (int i = 0; i < 2; ++i) {
      int idx = tid + i * 128;
      int row = idx >> 3, c8 = (idx & 7) * 8;
      uint4 vv = *(const uint4*)(qkv + baseV + (size_t)(kt * 32 + row) * rs + c8);
      const unsigned short* s = (const unsigned short*)&vv;
#pragma unroll
      for (int j = 0; j < 8; ++j) lVt[(c8 + j) * 40 + row] = s[j];
    }
    async_wait0();
    __syncthreads();

    // Waves whose q-rows are entirely below this key chunk skip the math.
    if (kt * 32 <= q0 + w * 16 + 15) {
      // S = Q @ K^T (16 q x 32 keys): 2 N-subtiles, D split into 2 K=32 chunks.
      AccF sacc[2];
#pragma unroll
      for (int ns = 0; ns < 2; ++ns) {
#pragma unroll
        for (int r = 0; r < 8; ++r) sacc[ns].f[r] = 0.0f;
#pragma unroll
        for (int kc = 0; kc < 2; ++kc) {
          bf16x16 aq = frag_a(&lQ[(w * 16) * 72], 72, kc * 32);
          bf16x16 bk = frag_b(&lK[(ns * 16) * 72], 72, kc * 32);
          sacc[ns].v = wmma_bf16f32(aq, bk, sacc[ns].v);
        }
      }
      // Scale + causal mask.
#pragma unroll
      for (int ns = 0; ns < 2; ++ns)
#pragma unroll
        for (int r = 0; r < 8; ++r) {
          int gq = q0 + w * 16 + r + hi * 8;
          int gk = kt * 32 + ns * 16 + cn;
          float sv = sacc[ns].f[r] * scale;
          sacc[ns].f[r] = (gk <= gq) ? sv : -3.0e38f;
        }
      // Row max across the 16-lane half (row spread over N=lane&15).
      float rm[8];
#pragma unroll
      for (int r = 0; r < 8; ++r) rm[r] = fmaxf(sacc[0].f[r], sacc[1].f[r]);
#pragma unroll
      for (int m = 8; m >= 1; m >>= 1)
#pragma unroll
        for (int r = 0; r < 8; ++r) rm[r] = fmaxf(rm[r], __shfl_xor(rm[r], m, 32));

      float fac[8];
#pragma unroll
      for (int r = 0; r < 8; ++r) {
        float nm = fmaxf(row_max[r], rm[r]);
        fac[r] = __expf(row_max[r] - nm);
        row_max[r] = nm;
      }
      // P = exp(S - max); row sums.
      float psum[8];
#pragma unroll
      for (int r = 0; r < 8; ++r) psum[r] = 0.0f;
#pragma unroll
      for (int ns = 0; ns < 2; ++ns)
#pragma unroll
        for (int r = 0; r < 8; ++r) {
          float p = __expf(sacc[ns].f[r] - row_max[r]);
          sacc[ns].f[r] = p;
          psum[r] += p;
        }
#pragma unroll
      for (int m = 8; m >= 1; m >>= 1)
#pragma unroll
        for (int r = 0; r < 8; ++r) psum[r] += __shfl_xor(psum[r], m, 32);
#pragma unroll
      for (int r = 0; r < 8; ++r) row_sum[r] = row_sum[r] * fac[r] + psum[r];
      // Rescale running O.
#pragma unroll
      for (int d = 0; d < 4; ++d)
#pragma unroll
        for (int r = 0; r < 8; ++r) o[d].f[r] *= fac[r];

      // Stage P (C-layout -> A-layout) through per-wave LDS.
      unsigned short* pp = &lP[w][0];
#pragma unroll
      for (int ns = 0; ns < 2; ++ns)
#pragma unroll
        for (int r = 0; r < 8; ++r)
          pp[(r + hi * 8) * 40 + ns * 16 + cn] = f32_to_bf16(sacc[ns].f[r]);
      asm volatile("s_wait_dscnt 0" ::: "memory");  // RAW through LDS (same wave)

      // O += P @ V : A = P (16x32), B = V (32 keys x 64 d), via lVt [d][key].
      bf16x16 ap = frag_a(pp, 40, 0);
#pragma unroll
      for (int d = 0; d < 4; ++d) {
        bf16x16 bv = frag_b(&lVt[(d * 16) * 40], 40, 0);
        o[d].v = wmma_bf16f32(ap, bv, o[d].v);
      }
    }
  }

  // Normalize and emit y in bf16 (feeds output-projection GEMM A-matrix).
  const size_t baseY = ((size_t)b * SEQT) * CDIM + h * HDIM;
#pragma unroll
  for (int r = 0; r < 8; ++r) row_sum[r] = 1.0f / row_sum[r];
#pragma unroll
  for (int d = 0; d < 4; ++d) {
    size_t off = baseY + (size_t)(q0 + w * 16 + hi * 8) * CDIM + d * 16 + cn;
#pragma unroll
    for (int r = 0; r < 8; ++r, off += CDIM)
      y[off] = f32_to_bf16(o[d].f[r] * row_sum[r]);
  }
}

// ---------------------------------------------------------------------------
extern "C" void kernel_launch(void* const* d_in, const int* in_sizes, int n_in,
                              void* d_out, int out_size, void* d_ws, size_t ws_size,
                              hipStream_t stream) {
  const float* x      = (const float*)d_in[0];
  const float* W_attn = (const float*)d_in[1];
  const float* b_attn = (const float*)d_in[2];
  const float* W_proj = (const float*)d_in[3];
  const float* b_proj = (const float*)d_in[4];
  float* out = (float*)d_out;

  const int M = NBATCH * SEQT;        // 8192
  unsigned short* ws   = (unsigned short*)d_ws;
  unsigned short* xb   = ws;                                   // M*C
  unsigned short* wab  = xb  + (size_t)M * CDIM;               // 3C*C (W_attn^T)
  unsigned short* wpb  = wab + (size_t)CDIM * N3DIM;           // C*C  (W_proj^T)
  unsigned short* qkvb = wpb + (size_t)CDIM * CDIM;            // M*3C
  unsigned short* yb   = qkvb + (size_t)M * N3DIM;             // M*C

  // 1) f32 -> bf16 conversions (activations plain, weights transposed).
  {
    int n4 = (M * CDIM) / 4;
    cvt_f32_to_bf16<<<(n4 + 255) / 256, 256, 0, stream>>>(x, xb, n4);
    transpose_cvt_f32_to_bf16<<<dim3(N3DIM / 32, CDIM / 32), 256, 0, stream>>>(
        W_attn, wab, CDIM, N3DIM);
    transpose_cvt_f32_to_bf16<<<dim3(CDIM / 32, CDIM / 32), 256, 0, stream>>>(
        W_proj, wpb, CDIM, CDIM);
  }
  // 2) qkv = x @ W_attn + b_attn  (bf16 out).
  gemm_bias_wmma<true><<<dim3(N3DIM / 128, M / 128), 256, 0, stream>>>(
      xb, wab, b_attn, qkvb, M, N3DIM, CDIM);
  // 3) Causal flash attention per (qtile, head, batch).
  flash_attn_wmma<<<dim3(SEQT / 64, NHEADS, NBATCH), 128, 0, stream>>>(qkvb, yb);
  // 4) out = y @ W_proj + b_proj  (f32 out).
  gemm_bias_wmma<false><<<dim3(CDIM / 128, M / 128), 256, 0, stream>>>(
      yb, wpb, b_proj, out, M, CDIM, CDIM);
}
